// MultiHeadAttention_31525059953146
// MI455X (gfx1250) — matrix operand order repaired and checked by reading, not compiled, not run on hardware
//
#include <hip/hip_runtime.h>
#include <hip/hip_bf16.h>

// MI455X (gfx1250) multi-head causal attention, bf16 WMMA pipeline:
//   1. cvt x -> bf16; Wq/Wk/Wv/Wo -> bf16 transposed ([N][K]).
//   2. QKV projection GEMM: 128x64 tile / block, K-chunk 64, LDS double
//      buffer filled with GLOBAL_LOAD_ASYNC_TO_LDS_B128 (ASYNCcnt); WMMA fed
//      from batched ds_load_b128 clauses. RoPE fused in the epilogue.
//      Q,K stored [B,H,S,D]; V stored [B,H,D,S].
//   3. flash attention: per-wave 16 query rows, 32 keys/step, batched frag
//      loads -> back-to-back WMMAs, online softmax, P transposed through LDS,
//      global_prefetch of the next key block.
//   4. output projection GEMM (same LDS pipeline) -> fp32 d_out.

typedef __attribute__((ext_vector_type(16))) __bf16      v16bf;
typedef __attribute__((ext_vector_type(8)))  float        v8f;
typedef __attribute__((ext_vector_type(8)))  unsigned int v8u;
typedef __attribute__((ext_vector_type(4)))  unsigned int v4u;
typedef __attribute__((ext_vector_type(4)))  int          v4i;

#define NHEADS 16
#define HDIM   64
#define SEQ    2048
#define BATCH  2
#define EMB    1024
#define MTOT   (BATCH * SEQ)   // 4096 tokens

#if defined(__HIP_DEVICE_COMPILE__) &&                                   \
    __has_builtin(__builtin_amdgcn_global_load_async_to_lds_b128) &&     \
    __has_builtin(__builtin_amdgcn_s_wait_asynccnt)
#define USE_ASYNC_LDS 1
#else
#define USE_ASYNC_LDS 0
#endif

__device__ __forceinline__ unsigned short f2bf(float f) {
  unsigned int u = __builtin_bit_cast(unsigned int, f);
  u += 0x7FFFu + ((u >> 16) & 1u);           // round-to-nearest-even
  return (unsigned short)(u >> 16);
}

__device__ __forceinline__ v8u load_frag(const unsigned short* lo,
                                         const unsigned short* hi) {
  v4u a = *(const v4u*)lo;
  v4u b = *(const v4u*)hi;
  v8u r;
  r[0] = a[0]; r[1] = a[1]; r[2] = a[2]; r[3] = a[3];
  r[4] = b[0]; r[5] = b[1]; r[6] = b[2]; r[7] = b[3];
  return r;
}

__device__ __forceinline__ v8f vzero8() {
  v8f z;
#pragma unroll
  for (int i = 0; i < 8; ++i) z[i] = 0.0f;
  return z;
}

__device__ __forceinline__ v8f wmma_bf16(v8u a, v8u b, v8f c) {
  return __builtin_amdgcn_wmma_f32_16x16x32_bf16(
      false, __builtin_bit_cast(v16bf, a),
      false, __builtin_bit_cast(v16bf, b),
      (short)0, c, false, false);
}

// 16-byte global -> LDS copy (async when available)
__device__ __forceinline__ void copy16(const unsigned short* g,
                                       unsigned short* l) {
#if USE_ASYNC_LDS
  v4i* gg = (v4i*)g;                         // drop const, retype
  v4i* ll = (v4i*)l;
  __builtin_amdgcn_global_load_async_to_lds_b128(
      (__attribute__((address_space(1))) v4i*)gg,
      (__attribute__((address_space(3))) v4i*)ll, 0, 0);
#else
  *(v4u*)l = *(const v4u*)g;
#endif
}

__device__ __forceinline__ void async_wait0() {
#if USE_ASYNC_LDS
  __builtin_amdgcn_s_wait_asynccnt(0);
#endif
}

// stage a 128-row x 64-col bf16 tile (row stride EMB) into LDS, 256 threads
__device__ __forceinline__ void stage_A(const unsigned short* g,
                                        unsigned short* l, int tid) {
#pragma unroll
  for (int j = 0; j < 4; ++j) {
    int c = tid + j * 256;                   // 0..1023 16B chunks
    copy16(g + (c >> 3) * EMB + (c & 7) * 8, l + c * 8);
  }
}

// stage a 64-row x 64-col bf16 tile
__device__ __forceinline__ void stage_B(const unsigned short* g,
                                        unsigned short* l, int tid) {
#pragma unroll
  for (int j = 0; j < 2; ++j) {
    int c = tid + j * 256;                   // 0..511
    copy16(g + (c >> 3) * EMB + (c & 7) * 8, l + c * 8);
  }
}

// per-wave 16x64 MACC over one K-chunk of 64 from LDS tiles.
// All 10 fragment loads are issued first (single dscnt wait), then the
// 8 WMMAs go back-to-back into the XDL pipe.
__device__ __forceinline__ void compute_chunk(const unsigned short* lA,
                                              const unsigned short* lB,
                                              int wave, int lrow, int hi16,
                                              v8f acc[4]) {
  const unsigned short* ar = lA + (wave * 16 + lrow) * 64 + hi16 * 8;
  v8u a[2], b[8];
#pragma unroll
  for (int ks = 0; ks < 2; ++ks)
    a[ks] = load_frag(ar + ks * 32, ar + ks * 32 + 16);
#pragma unroll
  for (int ks = 0; ks < 2; ++ks)
#pragma unroll
    for (int ct = 0; ct < 4; ++ct) {
      const unsigned short* bp =
          lB + (ct * 16 + lrow) * 64 + hi16 * 8 + ks * 32;
      b[ks * 4 + ct] = load_frag(bp, bp + 16);
    }
#pragma unroll
  for (int ks = 0; ks < 2; ++ks)
#pragma unroll
    for (int ct = 0; ct < 4; ++ct)
      acc[ct] = wmma_bf16(a[ks], b[ks * 4 + ct], acc[ct]);
}

// 128x64 output tile per block: LDS double-buffered K-chunks of 64
__device__ __forceinline__ void gemm_mainloop(const unsigned short* Ag,
                                              const unsigned short* Bg,
                                              unsigned short* lds, int tid,
                                              v8f acc[4]) {
  const int lane = tid & 31;
  const int wave = tid >> 5;
  const int lrow = lane & 15;
  const int hi16 = (lane >> 4) & 1;
  unsigned short* A0 = lds;                  // 128*64
  unsigned short* B0 = A0 + 128 * 64;        // 64*64
  unsigned short* A1 = B0 + 64 * 64;
  unsigned short* B1 = A1 + 128 * 64;

#if USE_ASYNC_LDS
  stage_A(Ag, A0, tid);
  stage_B(Bg, B0, tid);
  async_wait0();
  __syncthreads();
  for (int kk = 0; kk < EMB; kk += 64) {
    const int ping = (kk >> 6) & 1;
    unsigned short* cA = ping ? A1 : A0;
    unsigned short* cB = ping ? B1 : B0;
    unsigned short* nA = ping ? A0 : A1;
    unsigned short* nB = ping ? B0 : B1;
    if (kk + 64 < EMB) {                     // issue next chunk while computing
      stage_A(Ag + kk + 64, nA, tid);
      stage_B(Bg + kk + 64, nB, tid);
    }
    compute_chunk(cA, cB, wave, lrow, hi16, acc);
    async_wait0();
    __syncthreads();
  }
#else
  for (int kk = 0; kk < EMB; kk += 64) {
    __syncthreads();
    stage_A(Ag + kk, A0, tid);
    stage_B(Bg + kk, B0, tid);
    __syncthreads();
    compute_chunk(A0, B0, wave, lrow, hi16, acc);
  }
#endif
}

// ---------------------------------------------------------------- conversions
__global__ void cvt_x_kernel(const float* __restrict__ x,
                             unsigned short* __restrict__ xb, int n) {
  int i = blockIdx.x * blockDim.x + threadIdx.x;
  if (i < n) xb[i] = f2bf(x[i]);
}

// W [K=1024][N=1024] fp32 -> WT [N][K] bf16
__global__ void cvt_wT_kernel(const float* __restrict__ W,
                              unsigned short* __restrict__ WT) {
  int i = blockIdx.x * blockDim.x + threadIdx.x;  // 0 .. 1048575
  int n = i >> 10, k = i & 1023;
  WT[i] = f2bf(W[k * EMB + n]);
}

// ------------------------------------------------------- QKV projection + RoPE
// grid: (MTOT/128, NHEADS, 3), block 256 (8 waves, 16 rows each x 64 cols)
__global__ __launch_bounds__(256) void qkv_proj_kernel(
    const unsigned short* __restrict__ xb,
    const unsigned short* __restrict__ WTq,
    const unsigned short* __restrict__ WTk,
    const unsigned short* __restrict__ WTv,
    unsigned short* __restrict__ Q,
    unsigned short* __restrict__ K,
    unsigned short* __restrict__ V) {
  __shared__ __align__(16) unsigned short lds[2 * (128 * 64 + 64 * 64)];
  const int tid  = threadIdx.x;
  const int lane = tid & 31;
  const int wave = tid >> 5;
  const int lrow = lane & 15;
  const int hi16 = (lane >> 4) & 1;
  const int mat  = blockIdx.z;                       // 0=Q 1=K 2=V
  const unsigned short* WT = (mat == 0) ? WTq : ((mat == 1) ? WTk : WTv);
  const int r0 = blockIdx.x * 128;
  const int h  = blockIdx.y;                         // head == 64-col tile
  const int c0 = h * HDIM;

  v8f acc[4];
#pragma unroll
  for (int ct = 0; ct < 4; ++ct) acc[ct] = vzero8();

  gemm_mainloop(xb + r0 * EMB, WT + c0 * EMB, lds, tid, acc);

  // epilogue: RoPE for Q/K, transpose-store for V
  const int r0w = r0 + wave * 16;
#pragma unroll
  for (int g = 0; g < 8; ++g) {
    const int gr = r0w + g + hi16 * 8;               // global token row
    const int b  = gr >> 11;                         // / SEQ
    const int s  = gr & (SEQ - 1);
    if (mat < 2) {
      unsigned short* dst = (mat == 0) ? Q : K;
      const float fs = (float)s;
#pragma unroll
      for (int ct = 0; ct < 4; ++ct) {
        const int d = ct * 16 + lrow;                // dim within head
        // inv_freq = 10000^{-(2i/D)}, 2i = d & ~1 ; log2(10000)=13.28771238
        float invf = exp2f(-((float)(d & ~1) * (1.0f / 64.0f)) *
                           13.287712379549449f);
        float ang = fs * invf;
        float cs = __cosf(ang), sn = __sinf(ang);
        float v  = acc[ct][g];
        float pv = __shfl_xor(v, 1, 32);             // RoPE pair partner
        float o  = (lane & 1) ? (v * cs + pv * sn) : (v * cs - pv * sn);
        dst[((b * NHEADS + h) * SEQ + s) * HDIM + d] = f2bf(o);
      }
    } else {
#pragma unroll
      for (int ct = 0; ct < 4; ++ct) {
        const int d = ct * 16 + lrow;
        V[((b * NHEADS + h) * HDIM + d) * SEQ + s] = f2bf(acc[ct][g]);
      }
    }
  }
}

// ------------------------------------------------------------ flash attention
// grid: (SEQ/128, BATCH*NHEADS), block 256 (8 waves x 16 query rows each)
__global__ __launch_bounds__(256) void attn_kernel(
    const unsigned short* __restrict__ Q,    // [B,H,S,D]
    const unsigned short* __restrict__ K,    // [B,H,S,D]
    const unsigned short* __restrict__ VT,   // [B,H,D,S]
    unsigned short* __restrict__ O) {        // [B*S, H*D] token-major
  __shared__ __align__(16) unsigned short plds[8 * 16 * 32];  // per-wave P
  const int lane = threadIdx.x & 31;
  const int wave = threadIdx.x >> 5;
  const int lrow = lane & 15;
  const int hi16 = (lane >> 4) & 1;
  const int bh   = blockIdx.y;
  const int q0   = blockIdx.x * 128 + wave * 16;

  const unsigned short* Qb = Q  + bh * SEQ * HDIM;
  const unsigned short* Kb = K  + bh * SEQ * HDIM;
  const unsigned short* Vb = VT + bh * HDIM * SEQ;

  // hold all 16 query rows x 64 dims in registers (2 K-chunks of 32)
  const unsigned short* qrow = Qb + (q0 + lrow) * HDIM + hi16 * 8;
  v8u qa[2];
#pragma unroll
  for (int c = 0; c < 2; ++c)
    qa[c] = load_frag(qrow + c * 32, qrow + c * 32 + 16);

  v8f o[4];
  float m[8], lsum[8];
#pragma unroll
  for (int ct = 0; ct < 4; ++ct) o[ct] = vzero8();
#pragma unroll
  for (int g = 0; g < 8; ++g) { m[g] = -1e30f; lsum[g] = 0.0f; }

  unsigned short* myp = plds + wave * (16 * 32);

  for (int kt = 0; kt <= q0 + 15; kt += 32) {
    // prefetch next key block (global_prefetch_b8)
    __builtin_prefetch(Kb + (kt + 32 + lrow) * HDIM, 0, 0);

    // batch all 4 K fragments, then 4 back-to-back WMMAs
    v8u kf[4];
#pragma unroll
    for (int t = 0; t < 2; ++t)
#pragma unroll
      for (int c = 0; c < 2; ++c) {
        const unsigned short* krow =
            Kb + (kt + t * 16 + lrow) * HDIM + c * 32 + hi16 * 8;
        kf[t * 2 + c] = load_frag(krow, krow + 16);
      }
    v8f sc[2];
    sc[0] = vzero8(); sc[1] = vzero8();
    sc[0] = wmma_bf16(qa[0], kf[0], sc[0]);
    sc[0] = wmma_bf16(qa[1], kf[1], sc[0]);
    sc[1] = wmma_bf16(qa[0], kf[2], sc[1]);
    sc[1] = wmma_bf16(qa[1], kf[3], sc[1]);

    const int kc0 = kt + lrow;
    const int kc1 = kt + 16 + lrow;
#pragma unroll
    for (int g = 0; g < 8; ++g) {
      const int qr = q0 + g + hi16 * 8;
      float s0 = sc[0][g] * 0.125f; if (kc0 > qr) s0 = -1e30f;  // 1/sqrt(64)
      float s1 = sc[1][g] * 0.125f; if (kc1 > qr) s1 = -1e30f;
      // row max across the 16 lanes of this half
      float red = fmaxf(s0, s1);
      red = fmaxf(red, __shfl_xor(red, 1, 32));
      red = fmaxf(red, __shfl_xor(red, 2, 32));
      red = fmaxf(red, __shfl_xor(red, 4, 32));
      red = fmaxf(red, __shfl_xor(red, 8, 32));
      float mnew = fmaxf(m[g], red);
      float corr = __expf(m[g] - mnew);
      m[g] = mnew;
      float p0 = __expf(s0 - mnew);
      float p1 = __expf(s1 - mnew);
      float rs = p0 + p1;
      rs += __shfl_xor(rs, 1, 32);
      rs += __shfl_xor(rs, 2, 32);
      rs += __shfl_xor(rs, 4, 32);
      rs += __shfl_xor(rs, 8, 32);
      lsum[g] = lsum[g] * corr + rs;
#pragma unroll
      for (int ct = 0; ct < 4; ++ct) o[ct][g] *= corr;
      // stage P (C-layout) into LDS for the A-layout re-read
      const int r = g + hi16 * 8;
      myp[r * 32 + lrow]      = f2bf(p0);
      myp[r * 32 + 16 + lrow] = f2bf(p1);
    }

    // P as 16x32 A-fragment (per-wave LDS region, in-order DS ops)
    const unsigned short* prow = myp + lrow * 32 + hi16 * 8;
    v8u pa = load_frag(prow, prow + 16);

    // batch all 4 V fragments, then 4 back-to-back WMMAs
    v8u vf[4];
#pragma unroll
    for (int ct = 0; ct < 4; ++ct) {
      const unsigned short* vrow =
          Vb + (ct * 16 + lrow) * SEQ + kt + hi16 * 8;
      vf[ct] = load_frag(vrow, vrow + 16);
    }
#pragma unroll
    for (int ct = 0; ct < 4; ++ct) o[ct] = wmma_bf16(pa, vf[ct], o[ct]);
  }

  const int h = bh & (NHEADS - 1);
  const int b = bh >> 4;
#pragma unroll
  for (int g = 0; g < 8; ++g) {
    const float inv = 1.0f / lsum[g];
    const int s   = q0 + g + hi16 * 8;
    const int tok = b * SEQ + s;
#pragma unroll
    for (int ct = 0; ct < 4; ++ct)
      O[tok * EMB + h * HDIM + ct * 16 + lrow] = f2bf(o[ct][g] * inv);
  }
}

// --------------------------------------------------------- output projection
// grid: (MTOT/128, EMB/64), block 256; fp32 output to d_out.
__global__ __launch_bounds__(256) void out_proj_kernel(
    const unsigned short* __restrict__ A,     // attn out [M][1024] bf16
    const unsigned short* __restrict__ WoT,   // [E][HD] bf16
    float* __restrict__ out) {
  __shared__ __align__(16) unsigned short lds[2 * (128 * 64 + 64 * 64)];
  const int tid  = threadIdx.x;
  const int lane = tid & 31;
  const int wave = tid >> 5;
  const int lrow = lane & 15;
  const int hi16 = (lane >> 4) & 1;
  const int r0 = blockIdx.x * 128;
  const int c0 = blockIdx.y * 64;

  v8f acc[4];
#pragma unroll
  for (int ct = 0; ct < 4; ++ct) acc[ct] = vzero8();

  gemm_mainloop(A + r0 * EMB, WoT + c0 * EMB, lds, tid, acc);

  const int r0w = r0 + wave * 16;
#pragma unroll
  for (int g = 0; g < 8; ++g) {
    const int row = r0w + g + hi16 * 8;
#pragma unroll
    for (int ct = 0; ct < 4; ++ct)
      out[row * EMB + c0 + ct * 16 + lrow] = acc[ct][g];
  }
}

// ---------------------------------------------------------------------- host
extern "C" void kernel_launch(void* const* d_in, const int* in_sizes, int n_in,
                              void* d_out, int out_size, void* d_ws,
                              size_t ws_size, hipStream_t stream) {
  const float* x  = (const float*)d_in[0];
  const float* Wq = (const float*)d_in[1];
  const float* Wk = (const float*)d_in[2];
  const float* Wv = (const float*)d_in[3];
  const float* Wo = (const float*)d_in[4];
  float* out = (float*)d_out;

  // workspace layout (bf16 = unsigned short), all 16B aligned
  char* ws = (char*)d_ws;
  const size_t XB   = (size_t)MTOT * EMB * 2;      // 8 MB
  const size_t WT1  = (size_t)EMB * EMB * 2;       // 2 MB
  const size_t QKV  = (size_t)MTOT * EMB * 2;      // 8 MB each
  unsigned short* xb  = (unsigned short*)(ws);
  unsigned short* WqT = (unsigned short*)(ws + XB);
  unsigned short* WkT = (unsigned short*)(ws + XB + WT1);
  unsigned short* WvT = (unsigned short*)(ws + XB + 2 * WT1);
  unsigned short* WoT = (unsigned short*)(ws + XB + 3 * WT1);
  unsigned short* Qb  = (unsigned short*)(ws + XB + 4 * WT1);
  unsigned short* Kb  = (unsigned short*)(ws + XB + 4 * WT1 + QKV);
  unsigned short* Vb  = (unsigned short*)(ws + XB + 4 * WT1 + 2 * QKV);
  unsigned short* Ob  = (unsigned short*)(ws + XB + 4 * WT1 + 3 * QKV);

  const int nx = MTOT * EMB;  // 4194304
  cvt_x_kernel<<<(nx + 255) / 256, 256, 0, stream>>>(x, xb, nx);
  cvt_wT_kernel<<<(EMB * EMB) / 256, 256, 0, stream>>>(Wq, WqT);
  cvt_wT_kernel<<<(EMB * EMB) / 256, 256, 0, stream>>>(Wk, WkT);
  cvt_wT_kernel<<<(EMB * EMB) / 256, 256, 0, stream>>>(Wv, WvT);
  cvt_wT_kernel<<<(EMB * EMB) / 256, 256, 0, stream>>>(Wo, WoT);

  qkv_proj_kernel<<<dim3(MTOT / 128, NHEADS, 3), 256, 0, stream>>>(
      xb, WqT, WkT, WvT, Qb, Kb, Vb);

  attn_kernel<<<dim3(SEQ / 128, BATCH * NHEADS), 256, 0, stream>>>(
      Qb, Kb, Vb, Ob);

  out_proj_kernel<<<dim3(MTOT / 128, EMB / 64), 256, 0, stream>>>(Ob, WoT, out);
}
